// StrideHDClassifier_68762426409220
// MI455X (gfx1250) — 8-bit matrix operand order repaired and checked by reading, compiled, not run on hardware
//
#include <hip/hip_runtime.h>


typedef __attribute__((ext_vector_type(4)))  float v4f;
typedef __attribute__((ext_vector_type(4)))  int   v4i;
typedef __attribute__((ext_vector_type(8)))  float v8f;
typedef __attribute__((ext_vector_type(16))) int   v16i;

typedef __attribute__((address_space(1))) v4i as1_v4i;
typedef __attribute__((address_space(3))) v4i as3_v4i;

#define B_DIM   8192
#define D_DIM   10240
#define C_DIM   256
#define BK      128
#define THREADS 64
#define BM      32
#define ROW_DW  36
#define NITER   (D_DIM / BK)
#define CHUNK_BYTES (C_DIM * BK)

#if __has_builtin(__builtin_amdgcn_global_load_async_to_lds_b128) && \
    __has_builtin(__builtin_amdgcn_s_wait_asynccnt)
#define HAVE_ASYNC_LDS 1
#else
#define HAVE_ASYNC_LDS 0
#endif

__device__ __forceinline__ unsigned pack4_fp8(v4f f) {
  unsigned u0 = __builtin_bit_cast(unsigned, f.x);
  unsigned u1 = __builtin_bit_cast(unsigned, f.y);
  unsigned u2 = __builtin_bit_cast(unsigned, f.z);
  unsigned u3 = __builtin_bit_cast(unsigned, f.w);
  unsigned t01 = __builtin_amdgcn_perm(u1, u0, 0x00000703u);
  unsigned t23 = __builtin_amdgcn_perm(u3, u2, 0x00000703u);
  unsigned r   = __builtin_amdgcn_perm(t23, t01, 0x05040100u);
  return r & 0x38383838u;
}

__device__ __forceinline__ void stage_row_fp8(int* dst, const float* __restrict__ src) {
#pragma unroll
  for (int j = 0; j < 8; ++j) {
    v4f f0 = *(const v4f*)(src + 16 * j);
    v4f f1 = *(const v4f*)(src + 16 * j + 4);
    v4f f2 = *(const v4f*)(src + 16 * j + 8);
    v4f f3 = *(const v4f*)(src + 16 * j + 12);
    v4i d;
    d.x = (int)pack4_fp8(f0);
    d.y = (int)pack4_fp8(f1);
    d.z = (int)pack4_fp8(f2);
    d.w = (int)pack4_fp8(f3);
    *(v4i*)(dst + 4 * j) = d;
  }
}

__device__ __forceinline__ void copy16_to_lds(char* l, const unsigned char* g) {
#if HAVE_ASYNC_LDS
  __builtin_amdgcn_global_load_async_to_lds_b128(
      (as1_v4i*)g, (as3_v4i*)l, 0, 0);
#else
  *(v4i*)l = *(const v4i*)g;
#endif
}

__device__ __forceinline__ void wait_async_lds() {
#if HAVE_ASYNC_LDS
  __builtin_amdgcn_s_wait_asynccnt(0);
#endif
}

__global__ __launch_bounds__(256) void classes_to_fp8(
    const float* __restrict__ classes, unsigned char* __restrict__ ws) {
  const size_t t = (size_t)blockIdx.x * 256 + threadIdx.x;
  const size_t e = t * 16;
  const int chunk  = (int)(e >> 15);
  const int within = (int)(e & 32767);
  const int c = within >> 7;
  const int j = within & 127;
  const float* src = classes + (size_t)c * D_DIM + (size_t)chunk * BK + j;
  v4f f0 = __builtin_nontemporal_load((const v4f*)(src + 0));
  v4f f1 = __builtin_nontemporal_load((const v4f*)(src + 4));
  v4f f2 = __builtin_nontemporal_load((const v4f*)(src + 8));
  v4f f3 = __builtin_nontemporal_load((const v4f*)(src + 12));
  v4i d;
  d.x = (int)pack4_fp8(f0);
  d.y = (int)pack4_fp8(f1);
  d.z = (int)pack4_fp8(f2);
  d.w = (int)pack4_fp8(f3);
  *(v4i*)(ws + e) = d;
}

template <bool USE_WS>
__global__ __launch_bounds__(THREADS) void hd_classify_fp8_wmma(
    const float* __restrict__ samples,
    const float* __restrict__ classes_f32,
    const unsigned char* __restrict__ classes_fp8,
    float* __restrict__ out)
{
  __shared__ int ldsB[2][C_DIM * ROW_DW];

  const int tid  = threadIdx.x;
  const int wave = tid >> 5;
  const int lane = tid & 31;
  const int lm16 = lane & 15;
  const int lhi  = lane >> 4;

  const int rowBase = blockIdx.x * BM + wave * 16;
  const float* aRow = samples + (size_t)(rowBase + lm16) * D_DIM;

  v8f acc[16];
  {
    v8f z = {0.f, 0.f, 0.f, 0.f, 0.f, 0.f, 0.f, 0.f};
#pragma unroll
    for (int t = 0; t < 16; ++t) acc[t] = z;
  }

  auto stage = [&](int buf, int chunkIdx) {
    if (USE_WS) {
      const unsigned char* src = classes_fp8 + (size_t)chunkIdx * CHUNK_BYTES;
      char* base = (char*)&ldsB[buf][0];
#pragma unroll
      for (int i = 0; i < CHUNK_BYTES / 16 / THREADS; ++i) {
        const int p = tid + THREADS * i;
        copy16_to_lds(base + (p >> 3) * (ROW_DW * 4) + ((p & 7) << 4),
                      src + (size_t)p * 16);
      }
    } else {
#pragma unroll
      for (int r = 0; r < C_DIM / THREADS; ++r) {
        const int c = tid + THREADS * r;
        stage_row_fp8(&ldsB[buf][c * ROW_DW],
                      classes_f32 + (size_t)c * D_DIM + (size_t)chunkIdx * BK);
      }
    }
  };

  stage(0, 0);
  wait_async_lds();
  __syncthreads();

  for (int it = 0; it < NITER; ++it) {
    const int k0  = it * BK;
    const int cur = it & 1;

    if (it + 1 < NITER) stage(cur ^ 1, it + 1);

    v16i aF;
    {
      const float* ap = aRow + k0 + 16 * lhi;
#pragma unroll
      for (int r = 0; r < 8; ++r) {
        v4f f0 = __builtin_nontemporal_load((const v4f*)(ap + 32 * (r >> 1) + 8 * (r & 1)));
        v4f f1 = __builtin_nontemporal_load((const v4f*)(ap + 32 * (r >> 1) + 8 * (r & 1) + 4));
        aF[2 * r]     = (int)pack4_fp8(f0);
        aF[2 * r + 1] = (int)pack4_fp8(f1);
      }
    }

    const int* base = &ldsB[cur][0];
#pragma unroll
    for (int t = 0; t < 16; ++t) {
      const int* rp = base + (t * 16 + lm16) * ROW_DW + 4 * lhi;
      v4i c0 = *(const v4i*)(rp + 0);
      v4i c1 = *(const v4i*)(rp + 8);
      v4i c2 = *(const v4i*)(rp + 16);
      v4i c3 = *(const v4i*)(rp + 24);
      v16i bF = {c0.x, c0.y, c0.z, c0.w, c1.x, c1.y, c1.z, c1.w,
                 c2.x, c2.y, c2.z, c2.w, c3.x, c3.y, c3.z, c3.w};
      acc[t] = __builtin_amdgcn_wmma_f32_16x16x128_fp8_fp8(
                   aF, bF, (short)0, acc[t], false, false);
    }

    wait_async_lds();
    __syncthreads();
  }

#pragma unroll
  for (int t = 0; t < 16; ++t) {
#pragma unroll
    for (int r = 0; r < 8; ++r) {
      const int m = r + 8 * lhi;
      __builtin_nontemporal_store(
          acc[t][r], out + (size_t)(rowBase + m) * C_DIM + t * 16 + lm16);
    }
  }
}

extern "C" void kernel_launch(void* const* d_in, const int* in_sizes, int n_in,
                              void* d_out, int out_size, void* d_ws, size_t ws_size,
                              hipStream_t stream) {
  (void)in_sizes; (void)n_in; (void)out_size;
  const float* samples = (const float*)d_in[0];
  const float* classes = (const float*)d_in[1];
  float* out = (float*)d_out;

  const size_t need = (size_t)C_DIM * D_DIM;
  if (ws_size >= need) {
    unsigned char* ws = (unsigned char*)d_ws;
    classes_to_fp8<<<dim3((C_DIM * D_DIM / 16) / 256), dim3(256), 0, stream>>>(
        classes, ws);
    hd_classify_fp8_wmma<true><<<dim3(B_DIM / BM), dim3(THREADS), 0, stream>>>(
        samples, classes, ws, out);
  } else {
    hd_classify_fp8_wmma<false><<<dim3(B_DIM / BM), dim3(THREADS), 0, stream>>>(
        samples, classes, nullptr, out);
  }
}
